// VQVAE_83631603188440
// MI455X (gfx1250) — compile-verified, matrix operand order checked by reading, not run on hardware
//
#include <hip/hip_runtime.h>
#include <hip/hip_bf16.h>

typedef __attribute__((ext_vector_type(16))) _Float16 v16h;
typedef __attribute__((ext_vector_type(8)))  _Float16 h8;
typedef __attribute__((ext_vector_type(8)))  float    v8f;

union FragU { v16h v; h8 h[2]; };

__device__ __forceinline__ v16h load_frag16(const _Float16* p) {
    FragU f;
    f.h[0] = *reinterpret_cast<const h8*>(p);        // K = kh+0..7   (16B load)
    f.h[1] = *reinterpret_cast<const h8*>(p + 16);   // K = kh+16..23 (16B load)
    return f.v;
}

// ---------------------------------------------------------------------------
// WMMA GEMM, 2x2 register-blocked: one wave computes a 32x32 C tile.
// C[M,N] = A[M,K] * Bt[Npad,K]^T + bias, fused activation.
// A,Bt row-major f16, K multiple of 32 (16B-aligned rows).
// M multiple of 32; Npad multiple of 32 (zero-padded rows in Bt).
// Per k-step: 8x b128 loads feed 4 WMMAs (A/B fragments each reused twice).
// act: 0=none 1=relu 2=sigmoid
// ---------------------------------------------------------------------------
__global__ void k_gemm_wmma(const _Float16* __restrict__ A,
                            const _Float16* __restrict__ Bt,
                            const float* __restrict__ bias,
                            float* __restrict__ C,
                            int M, int N, int K, int tilesM, int tilesN, int act) {
    int wave = (int)((blockIdx.x * blockDim.x + threadIdx.x) >> 5);
    if (wave >= tilesM * tilesN) return;              // wave-uniform: EXEC stays all-1s
    int tm = wave / tilesN, tn = wave % tilesN;
    int lane = threadIdx.x & 31;
    int lrow = lane & 15;
    int kh   = (lane >> 4) << 3;                      // lanes16-31 hold K+8 / K+24 halves

    long m0 = (long)tm * 32, n0 = (long)tn * 32;
    const _Float16* a0 = A  + (m0 + lrow) * (long)K + kh;
    const _Float16* a1 = a0 + 16 * (long)K;
    const _Float16* b0 = Bt + (n0 + lrow) * (long)K + kh;
    const _Float16* b1 = b0 + 16 * (long)K;

    v8f acc00 = {}, acc01 = {}, acc10 = {}, acc11 = {};
    for (int k0 = 0; k0 < K; k0 += 32) {
        v16h af0 = load_frag16(a0 + k0);
        v16h af1 = load_frag16(a1 + k0);
        v16h bf0 = load_frag16(b0 + k0);
        v16h bf1 = load_frag16(b1 + k0);
        acc00 = __builtin_amdgcn_wmma_f32_16x16x32_f16(false, af0, false, bf0,
                                                       (short)0, acc00, false, false);
        acc01 = __builtin_amdgcn_wmma_f32_16x16x32_f16(false, af0, false, bf1,
                                                       (short)0, acc01, false, false);
        acc10 = __builtin_amdgcn_wmma_f32_16x16x32_f16(false, af1, false, bf0,
                                                       (short)0, acc10, false, false);
        acc11 = __builtin_amdgcn_wmma_f32_16x16x32_f16(false, af1, false, bf1,
                                                       (short)0, acc11, false, false);
    }

    int colL  = (int)n0 + lrow, colH = colL + 16;
    int rbase = (int)m0 + kh;                         // VGPR r -> row rbase+r (+8 hi lanes)
    float bvL = (bias && colL < N) ? bias[colL] : 0.0f;
    float bvH = (bias && colH < N) ? bias[colH] : 0.0f;

    auto stor = [&](const v8f& acc, int row0, int col, float bv) {
        if (col >= N) return;
        #pragma unroll
        for (int r = 0; r < 8; ++r) {
            int row = row0 + r;
            if (row < M) {
                float v = acc[r] + bv;
                if (act == 1)      v = v > 0.0f ? v : 0.0f;
                else if (act == 2) v = 1.0f / (1.0f + __expf(-v));
                C[(long)row * N + col] = v;
            }
        }
    };
    stor(acc00, rbase,      colL, bvL);
    stor(acc01, rbase,      colH, bvH);
    stor(acc10, rbase + 16, colL, bvL);
    stor(acc11, rbase + 16, colH, bvH);
}

// ---------------------------------------------------------------------------
// im2col for normal conv (f32 -> f16, zero-padded K tail).
// patch[m][k], m=(b,oy,ox), k=(ky*kw+kx)*C+ci
// ---------------------------------------------------------------------------
__global__ void k_im2col_conv(const float* __restrict__ x, _Float16* __restrict__ patch,
                              int B, int H, int W, int C, int khh, int kww, int stride,
                              int padt, int padl, int OH, int OW, int Kreal, int Kpad) {
    long total = (long)B * OH * OW * Kpad;
    long step = (long)gridDim.x * blockDim.x;
    for (long i = (long)blockIdx.x * blockDim.x + threadIdx.x; i < total; i += step) {
        int k = (int)(i % Kpad); long m = i / Kpad;
        float v = 0.0f;
        if (k < Kreal) {
            int ci = k % C; int kk = k / C; int kx = kk % kww; int ky = kk / kww;
            int ox = (int)(m % OW); long t = m / OW;
            int oy = (int)(t % OH); int b = (int)(t / OH);
            int iy = oy * stride - padt + ky, ix = ox * stride - padl + kx;
            if (iy >= 0 && iy < H && ix >= 0 && ix < W)
                v = x[(((long)b * H + iy) * W + ix) * C + ci];
        }
        patch[i] = (_Float16)v;
    }
}

// im2col for one parity class of a 4x4/s2 conv_transpose (2x2 taps, K=4*C).
__global__ void k_im2col_deconv(const float* __restrict__ x, _Float16* __restrict__ patch,
                                int B, int H, int W, int C,
                                int dy0, int dy1, int dx0, int dx1, int T, int U) {
    int Kpad = 4 * C;
    long total = (long)B * T * U * Kpad;
    long step = (long)gridDim.x * blockDim.x;
    for (long i = (long)blockIdx.x * blockDim.x + threadIdx.x; i < total; i += step) {
        int k = (int)(i % Kpad); long m = i / Kpad;
        int ci = k % C; int j = k / C;
        int dy = (j >> 1) ? dy1 : dy0;
        int dx = (j & 1)  ? dx1 : dx0;
        int u = (int)(m % U); long t0 = m / U;
        int t = (int)(t0 % T); int b = (int)(t0 / T);
        int iy = t + dy, ix = u + dx;
        float v = 0.0f;
        if (iy >= 0 && iy < H && ix >= 0 && ix < W)
            v = x[(((long)b * H + iy) * W + ix) * C + ci];
        patch[i] = (_Float16)v;
    }
}

// Weight prep: HWIO f32 -> Bt[Npad,Kpad] f16 with optional folded inference-BN.
__global__ void k_prep_w_conv(const float* __restrict__ w, const float* __restrict__ bias,
                              const float* __restrict__ g, const float* __restrict__ bb,
                              const float* __restrict__ mm, const float* __restrict__ vv,
                              _Float16* __restrict__ Bt, float* __restrict__ biasOut,
                              int C, int Co, int Npad, int Kreal, int Kpad) {
    long total = (long)Npad * Kpad;
    long step = (long)gridDim.x * blockDim.x;
    for (long i = (long)blockIdx.x * blockDim.x + threadIdx.x; i < total; i += step) {
        int k = (int)(i % Kpad); int co = (int)(i / Kpad);
        float scale = 1.0f, shift = 0.0f;
        if (co < Co) {
            if (g) {
                float s = rsqrtf(vv[co] + 1e-3f);
                scale = g[co] * s;
                shift = (bias[co] - mm[co]) * scale + bb[co];
            } else {
                shift = bias[co];
            }
        }
        float v = 0.0f;
        if (co < Co && k < Kreal) v = w[(long)k * Co + co] * scale;   // HWIO: flat = k*Co+co
        Bt[i] = (_Float16)v;
        if (k == 0) biasOut[co] = (co < Co) ? shift : 0.0f;
    }
}

__global__ void k_prep_w_deconv(const float* __restrict__ w, const float* __restrict__ bias,
                                _Float16* __restrict__ Bt, float* __restrict__ biasOut,
                                int Ci, int Co, int Npad,
                                int ky0, int ky1, int kx0, int kx1) {
    int Kpad = 4 * Ci;
    long total = (long)Npad * Kpad;
    long step = (long)gridDim.x * blockDim.x;
    for (long i = (long)blockIdx.x * blockDim.x + threadIdx.x; i < total; i += step) {
        int k = (int)(i % Kpad); int co = (int)(i / Kpad);
        int ci = k % Ci; int j = k / Ci;
        int ky = (j >> 1) ? ky1 : ky0;
        int kx = (j & 1)  ? kx1 : kx0;
        float v = (co < Co) ? w[((long)(ky * 4 + kx) * Ci + ci) * Co + co] : 0.0f;
        Bt[i] = (_Float16)v;
        if (k == 0) biasOut[co] = (co < Co) ? bias[co] : 0.0f;
    }
}

__global__ void k_scatter_parity(const float* __restrict__ Cp, float* __restrict__ out,
                                 int B, int T, int U, int Co, int ey, int ex) {
    long total = (long)B * T * U * Co;
    long step = (long)gridDim.x * blockDim.x;
    for (long i = (long)blockIdx.x * blockDim.x + threadIdx.x; i < total; i += step) {
        int co = (int)(i % Co); long m = i / Co;
        int u = (int)(m % U); long t0 = m / U;
        int t = (int)(t0 % T); int b = (int)(t0 / T);
        out[(((long)b * (2 * T) + (2 * t + ey)) * (2 * U) + (2 * u + ex)) * Co + co] = Cp[i];
    }
}

__global__ void k_add_relu(const float* a, const float* b, float* o, long n) {
    long step = (long)gridDim.x * blockDim.x;
    for (long i = (long)blockIdx.x * blockDim.x + threadIdx.x; i < n; i += step) {
        float v = a[i] + b[i];
        o[i] = v > 0.0f ? v : 0.0f;
    }
}

__global__ void k_copy(const float* a, float* o, long n) {
    long step = (long)gridDim.x * blockDim.x;
    for (long i = (long)blockIdx.x * blockDim.x + threadIdx.x; i < n; i += step) o[i] = a[i];
}

__global__ void k_zero(float* p, long n) {
    long step = (long)gridDim.x * blockDim.x;
    for (long i = (long)blockIdx.x * blockDim.x + threadIdx.x; i < n; i += step) p[i] = 0.0f;
}

// flat[n][d] with NCHW->(-1,64) mapping: n = ((b*128+c)*16 + hw/64), d = hw%64
__global__ void k_flat_f16(const float* __restrict__ ze, _Float16* __restrict__ flat) {
    long total = 32768L * 64;
    long step = (long)gridDim.x * blockDim.x;
    for (long i = (long)blockIdx.x * blockDim.x + threadIdx.x; i < total; i += step) {
        int n = (int)(i >> 6); int d = (int)(i & 63);
        int b = n >> 11; int c = (n >> 4) & 127;
        int hw = ((n & 15) << 6) + d; int h = hw >> 5; int w = hw & 31;
        flat[i] = (_Float16)ze[(((long)b * 32 + h) * 32 + w) * 128 + c];
    }
}

__global__ void k_emb_prep(const float* __restrict__ emb, _Float16* __restrict__ ef,
                           float* __restrict__ en2) {
    int k = blockIdx.x * blockDim.x + threadIdx.x;
    if (k >= 512) return;
    float s = 0.0f;
    for (int d = 0; d < 64; ++d) {
        float v = emb[k * 64 + d];
        ef[k * 64 + d] = (_Float16)v;
        s += v * v;
    }
    en2[k] = s;
}

__global__ void k_argmin(const float* __restrict__ G, const float* __restrict__ en2,
                         int* __restrict__ idx, float* __restrict__ idxf, int Nrows) {
    int n = blockIdx.x * blockDim.x + threadIdx.x;
    if (n >= Nrows) return;
    const float* g = G + (long)n * 512;
    float best = 3.4e38f; int bi = 0;
    for (int k = 0; k < 512; ++k) {
        float d = en2[k] - 2.0f * g[k];        // |f|^2 constant per row -> same argmin
        if (d < best) { best = d; bi = k; }
    }
    idx[n] = bi;
    idxf[n] = (float)bi;
}

__global__ void k_loss_accum(const float* __restrict__ ze, const float* __restrict__ emb,
                             const int* __restrict__ idx, float* __restrict__ acc) {
    long total = 32768L * 64;
    long step = (long)gridDim.x * blockDim.x;
    float p = 0.0f;
    for (long i = (long)blockIdx.x * blockDim.x + threadIdx.x; i < total; i += step) {
        int n = (int)(i >> 6); int d = (int)(i & 63);
        int b = n >> 11; int c = (n >> 4) & 127;
        int hw = ((n & 15) << 6) + d; int h = hw >> 5; int w = hw & 31;
        float zf = ze[(((long)b * 32 + h) * 32 + w) * 128 + c];
        float ev = emb[(long)idx[n] * 64 + d];
        float df = ev - zf;
        p += df * df;
    }
    atomicAdd(acc, p);
}

__global__ void k_loss_final(const float* acc, float* out) {
    if (blockIdx.x == 0 && threadIdx.x == 0)
        out[0] = 0.25f * acc[0] / (32768.0f * 64.0f);
}

// ---------------------------------------------------------------------------
extern "C" void kernel_launch(void* const* d_in, const int* in_sizes, int n_in,
                              void* d_out, int out_size, void* d_ws, size_t ws_size,
                              hipStream_t stream) {
    (void)in_sizes; (void)n_in; (void)out_size; (void)ws_size;
    auto F = [&](int i) { return (const float*)d_in[i]; };
    const float* X = F(0);
    float* OUT = (float*)d_out;
    // d_out layout: x_recon[3145728] | z_e[2097152] | zq[2097152] | idx[32768] | loss[1]
    float* outXr   = OUT;
    float* outZe   = OUT + 3145728;
    float* outZq   = OUT + 5242880;
    float* outIdx  = OUT + 7340032;
    float* outLoss = OUT + 7372800;

    // -------- workspace arena --------
    char* ws = (char*)d_ws;
    size_t off = 0;
    auto walloc = [&](size_t bytes) {
        void* p = ws + off;
        off = (off + bytes + 255) & ~(size_t)255;
        return p;
    };
    float*     actA  = (float*)walloc(8388608UL * 4);     // up to [16,128,128,32]
    float*     actB  = (float*)walloc(8388608UL * 4);
    float*     actC  = (float*)walloc(8388608UL * 4);
    float*     ze    = (float*)walloc(2097152UL * 4);     // [16,32,32,128]
    _Float16*  arena = (_Float16*)walloc(67108864UL);     // im2col / G overlay (64MB)
    _Float16*  flatF = (_Float16*)walloc(32768UL * 64 * 2);
    float*     Ctmp  = (float*)walloc(2097152UL * 4);     // parity GEMM output
    _Float16*  BtW   = (_Float16*)walloc(1UL << 20);      // weight tile (max 288KB)
    float*     biasW = (float*)walloc(512 * 4);
    _Float16*  embF  = (_Float16*)walloc(512 * 64 * 2);
    float*     en2   = (float*)walloc(512 * 4);
    int*       idxI  = (int*)walloc(32768 * 4);
    float*     lossA = (float*)walloc(256);

    auto gridFor = [](long n) {
        long b = (n + 255) / 256;
        if (b > 32768) b = 32768;
        if (b < 1) b = 1;
        return (unsigned)b;
    };

    // M multiple of 32 in all calls; Npad multiple of 32 (Bt zero-padded).
    auto gemm = [&](const _Float16* A, const _Float16* Bt, const float* bias, float* C,
                    int M, int N, int Npad, int K, int act) {
        int tilesM = (M + 31) / 32, tilesN = Npad / 32;
        long threadsTotal = (long)tilesM * tilesN * 32;
        unsigned blocks = (unsigned)((threadsTotal + 255) / 256);
        k_gemm_wmma<<<blocks, 256, 0, stream>>>(A, Bt, bias, C, M, N, K, tilesM, tilesN, act);
    };

    // Conv layer: im2col + weight-prep(+BN fold) + WMMA GEMM, out NHWC=[M,Co].
    auto convL = [&](const float* in, int H, int W, int C,
                     const float* w, const float* b,
                     const float* g, const float* bb, const float* mm, const float* vv,
                     int ksz, int stride, int pad, int OH, int OW,
                     int Co, int Npad, int Kpad, float* out, int act) {
        int Kreal = ksz * ksz * C;
        long M = 16L * OH * OW;
        k_prep_w_conv<<<gridFor((long)Npad * Kpad), 256, 0, stream>>>(
            w, b, g, bb, mm, vv, BtW, biasW, C, Co, Npad, Kreal, Kpad);
        k_im2col_conv<<<gridFor(M * Kpad), 256, 0, stream>>>(
            in, arena, 16, H, W, C, ksz, ksz, stride, pad, pad, OH, OW, Kreal, Kpad);
        gemm(arena, BtW, biasW, out, (int)M, Co, Npad, Kpad, act);
    };

    auto resblock = [&](const float* in, float* tmp, float* out, int base) {
        convL(in,  32, 32, 128, F(base + 0), F(base + 1), F(base + 2), F(base + 3),
              F(base + 4), F(base + 5), 3, 1, 1, 32, 32, 128, 128, 1152, tmp, 1);
        convL(tmp, 32, 32, 128, F(base + 6), F(base + 7), F(base + 8), F(base + 9),
              F(base + 10), F(base + 11), 3, 1, 1, 32, 32, 128, 128, 1152, out, 0);
        long n = 2097152;
        k_add_relu<<<gridFor(n), 256, 0, stream>>>(out, in, out, n);
    };

    // conv_transpose 4x4/s2 SAME via 4 parity sub-convs (oy=2t+ey: ky in {1,3}/{0,2}).
    auto deconv = [&](const float* in, int H, int W, int C,
                      const float* w, const float* b, int Co, int Npad,
                      float* out, int act) {
        int Kpad = 4 * C;
        int T = H, U = W;
        long M = 16L * T * U;
        for (int ey = 0; ey < 2; ++ey) {
            int ky0 = ey ? 0 : 1, ky1 = ey ? 2 : 3;
            int dy0 = ey ? 1 : 0, dy1 = ey ? 0 : -1;
            for (int ex = 0; ex < 2; ++ex) {
                int kx0 = ex ? 0 : 1, kx1 = ex ? 2 : 3;
                int dx0 = ex ? 1 : 0, dx1 = ex ? 0 : -1;
                k_prep_w_deconv<<<gridFor((long)Npad * Kpad), 256, 0, stream>>>(
                    w, b, BtW, biasW, C, Co, Npad, ky0, ky1, kx0, kx1);
                k_im2col_deconv<<<gridFor(M * Kpad), 256, 0, stream>>>(
                    in, arena, 16, H, W, C, dy0, dy1, dx0, dx1, T, U);
                gemm(arena, BtW, biasW, Ctmp, (int)M, Co, Npad, Kpad, act);
                k_scatter_parity<<<gridFor(M * Co), 256, 0, stream>>>(
                    Ctmp, out, 16, T, U, Co, ey, ex);
            }
        }
    };

    // ================= encoder =================
    convL(X,    256, 256,  3, F(1), F(2), nullptr, nullptr, nullptr, nullptr,
          4, 2, 1, 128, 128,  32,  32,   64, actA, 1);
    convL(actA, 128, 128, 32, F(3), F(4), nullptr, nullptr, nullptr, nullptr,
          4, 2, 1,  64,  64,  64,  64,  512, actB, 1);
    convL(actB,  64,  64, 64, F(5), F(6), nullptr, nullptr, nullptr, nullptr,
          4, 2, 1,  32,  32, 128, 128, 1024, actA, 1);
    resblock(actA, actB, actC, 7);    // res1
    resblock(actC, actB, ze,  19);    // res2 -> z_e

    // ================= vector quantizer =================
    k_emb_prep<<<2, 256, 0, stream>>>(F(31), embF, en2);
    k_flat_f16<<<gridFor(32768L * 64), 256, 0, stream>>>(ze, flatF);
    gemm(flatF, embF, nullptr, (float*)arena, 32768, 512, 512, 64, 0);   // G = flat @ emb^T
    k_argmin<<<(32768 + 255) / 256, 256, 0, stream>>>((float*)arena, en2, idxI, outIdx, 32768);
    k_zero<<<1, 256, 0, stream>>>(lossA, 1);
    k_loss_accum<<<256, 256, 0, stream>>>(ze, F(31), idxI, lossA);
    k_loss_final<<<1, 1, 0, stream>>>(lossA, outLoss);
    k_copy<<<gridFor(2097152), 256, 0, stream>>>(ze, outZe, 2097152);
    k_copy<<<gridFor(2097152), 256, 0, stream>>>(ze, outZq, 2097152);  // straight-through value == z_e

    // ================= decoder (consumes straight-through value = z_e) =====
    resblock(ze,   actA, actB, 32);   // res3
    resblock(actB, actA, actC, 44);   // res4
    deconv(actC, 32, 32, 128, F(56), F(57), 64, 64, actA, 1);   // -> [16,64,64,64]
    deconv(actA, 64, 64,  64, F(58), F(59), 32, 32, actB, 1);   // -> [16,128,128,32]
    deconv(actB, 128, 128, 32, F(60), F(61), 3, 32, outXr, 2);  // -> x_recon, sigmoid (Npad 32)
}
